// Coupling_69604239999415
// MI455X (gfx1250) — matrix operand order repaired and checked by reading, not compiled, not run on hardware
//
#include <hip/hip_runtime.h>
#include <math.h>
#include <stdint.h>

typedef __attribute__((ext_vector_type(16))) __bf16 v16bf;
typedef __attribute__((ext_vector_type(8)))  __bf16 v8bf;
typedef __attribute__((ext_vector_type(8)))  float  v8f;
typedef __attribute__((ext_vector_type(4)))  unsigned int v4u;
typedef __attribute__((ext_vector_type(8)))  int v8i;
typedef __attribute__((ext_vector_type(4)))  int v4i;

#define MTOT 16384   // B*H*W = 16*32*32
#define HW   1024

#if __has_builtin(__builtin_amdgcn_tensor_load_to_lds) && __has_builtin(__builtin_amdgcn_s_wait_tensorcnt)
#define USE_TDM 1
#else
#define USE_TDM 0
#endif

// ---------------- small device helpers ----------------
__device__ __forceinline__ float red_max16(float v) {
    #pragma unroll
    for (int m = 1; m < 16; m <<= 1) v = fmaxf(v, __shfl_xor(v, m, 32));
    return v;
}
__device__ __forceinline__ float red_sum16(float v) {
    #pragma unroll
    for (int m = 1; m < 16; m <<= 1) v += __shfl_xor(v, m, 32);
    return v;
}
__device__ __forceinline__ float safe_logf(float x) { return __logf(fmaxf(x, 1e-22f)); }

// ---------------- weight-norm prep: Wt[Npad][Kpad] bf16, k = tap*Cin+ci ----------------
__global__ void wnorm_kernel(const float* __restrict__ v, const float* __restrict__ g,
                             __bf16* __restrict__ Wt, int N, int Cin, int taps, int Kpad)
{
    __shared__ float red[64];
    const int o = blockIdx.x;      // 0..Npad-1
    const int t = threadIdx.x;     // 64 threads
    const int K = Cin * taps;
    float s = 0.f;
    if (o < N)
        for (int k = t; k < K; k += 64) { float x = v[(size_t)o * K + k]; s += x * x; }
    red[t] = s; __syncthreads();
    for (int st = 32; st > 0; st >>= 1) { if (t < st) red[t] += red[t + st]; __syncthreads(); }
    const float inv = (o < N) ? g[o] * rsqrtf(red[0]) : 0.f;
    for (int k = t; k < Kpad; k += 64) {
        float val = 0.f;
        if (o < N && k < K) {
            int ci = k % Cin, tp = k / Cin;                 // k = tap*Cin + ci
            val = v[(size_t)o * K + ci * taps + tp] * inv;  // OIHW flat
        }
        Wt[(size_t)o * Kpad + k] = (__bf16)val;
    }
}

// ---------------- generic implicit-GEMM conv, bf16 WMMA, f32 accum ----------------
// A: f32 [MTOT][Cin] (BHWC). taps: 1 (1x1) or 9 (3x3 pad 1). Wt: bf16 [Npad][Kpad].
// C: f32 [MTOT][N] = A_im2col @ Wt^T + bias
__global__ __launch_bounds__(256)
void gemm_conv_kernel(const float* __restrict__ A, int Cin, int taps,
                      const __bf16* __restrict__ Wt, const float* __restrict__ bias,
                      float* __restrict__ C, int N, int Kpad)
{
    __shared__ __bf16 As[64][32];
    __shared__ __bf16 Bs[64][32];
    const int K   = Cin * taps;
    const int m0  = blockIdx.x * 64;
    const int n0  = blockIdx.y * 64;
    const int tid = threadIdx.x;
    const int lane = tid & 31;
    const int wv   = tid >> 5;
    const int rowg = wv >> 1, colg = wv & 1;
    const int nl = lane & 15, hh = lane >> 4;
    const int lr = tid >> 2;          // 0..63 (row for A, n-row for B)
    const int lk = (tid & 3) * 8;     // 0,8,16,24

    const int m = m0 + lr;
    const int b = m >> 10, y = (m >> 5) & 31, x = m & 31;

    v8f acc0 = {}; v8f acc1 = {};
#if USE_TDM
    const unsigned bs_lds = (unsigned)(uintptr_t)(void*)&Bs[0][0];
#endif

    for (int k0 = 0; k0 < Kpad; k0 += 32) {
        // ---- stage B tile: 64x32 bf16 tile of Wt via Tensor Data Mover ----
#if USE_TDM
        if (wv == 0) {
            const unsigned long long ga =
                (unsigned long long)(uintptr_t)(Wt + (size_t)n0 * Kpad + k0);
            v4u g0;
            g0[0] = 1u;                                        // count=1, user mode
            g0[1] = bs_lds;                                    // lds_addr (bytes)
            g0[2] = (unsigned)(ga & 0xffffffffu);              // global_addr[31:0]
            g0[3] = (unsigned)((ga >> 32) & 0x1ffffffu) | (2u << 30); // addr[56:32] | type=2
            v8i g1;
            g1[0] = (int)(1u << 16);                           // mask=0, data_size=1 (2B)
            g1[1] = (int)0xFFFF0000u;                          // tensor_dim0 lo16 = 0xFFFF
            g1[2] = (int)(0x7FFFu | (0xFFFFu << 16));          // td0 hi=0x7FFF, td1 lo=0xFFFF
            g1[3] = (int)(0x7FFFu | (32u << 16));              // td1 hi=0x7FFF, tile_dim0=32
            g1[4] = 64;                                        // tile_dim1=64, tile_dim2=0
            g1[5] = Kpad;                                      // tensor_dim0_stride[31:0]
            g1[6] = 0; g1[7] = 0;
            v4i g2 = {0, 0, 0, 0}, g3 = {0, 0, 0, 0};
            v8i g4 = {0, 0, 0, 0, 0, 0, 0, 0};
            __builtin_amdgcn_tensor_load_to_lds(g0, g1, g2, g3, g4, 0);
        }
#else
        *(v8bf*)&Bs[lr][lk] = *(const v8bf*)(Wt + (size_t)(n0 + lr) * Kpad + k0 + lk);
#endif
        // ---- stage A tile (im2col, f32 -> bf16) ----
        {
            float va[8];
            if (taps == 1) {
                // contiguous channels; K == Kpad for all 1x1 layers
                const float* src = A + (size_t)m * Cin + k0 + lk;
                *(float4*)&va[0] = *(const float4*)src;
                *(float4*)&va[4] = *(const float4*)(src + 4);
            } else if ((Cin & 31) == 0) {
                // 3x3 with Cin % 32 == 0: one 32-K chunk never crosses a tap
                const int tp  = k0 / Cin;                  // uniform per chunk
                const int cib = k0 - tp * Cin + lk;        // channel base, contiguous
                const int yy = y + tp / 3 - 1, xx = x + tp % 3 - 1;
                if (yy >= 0 && yy < 32 && xx >= 0 && xx < 32) {
                    const float* src = A + (size_t)(((b << 5) + yy) * 32 + xx) * Cin + cib;
                    *(float4*)&va[0] = *(const float4*)src;
                    *(float4*)&va[4] = *(const float4*)(src + 4);
                } else {
                    #pragma unroll
                    for (int j = 0; j < 8; ++j) va[j] = 0.f;
                }
            } else {
                // general path (input conv: Cin=8, K=72, Kpad=96)
                #pragma unroll
                for (int j = 0; j < 8; ++j) {
                    const int kk = k0 + lk + j;
                    float val = 0.f;
                    if (kk < K) {
                        const int tp = kk / Cin, ci = kk - tp * Cin;
                        const int yy = y + tp / 3 - 1, xx = x + tp % 3 - 1;
                        if (yy >= 0 && yy < 32 && xx >= 0 && xx < 32)
                            val = A[(size_t)(((b << 5) + yy) * 32 + xx) * Cin + ci];
                    }
                    va[j] = val;
                }
            }
            v8bf pk;
            #pragma unroll
            for (int j = 0; j < 8; ++j) pk[j] = (__bf16)va[j];
            *(v8bf*)&As[lr][lk] = pk;
        }
#if USE_TDM
        if (wv == 0) __builtin_amdgcn_s_wait_tensorcnt((short)0);
#endif
        __syncthreads();
        // ---- A fragment (16x32 bf16, K(i,h) = (i<8?i:i+8)+8h) ----
        v16bf a;
        {
            const int row = rowg * 16 + nl;
            v8bf lo = *(const v8bf*)&As[row][8 * hh];
            v8bf hi = *(const v8bf*)&As[row][16 + 8 * hh];
            #pragma unroll
            for (int i = 0; i < 8; ++i) { a[i] = lo[i]; a[8 + i] = hi[i]; }
        }
        // ---- two B fragments (K(i,h) = (i<8?i:i+8)+8h, col = n) + WMMA ----
        #pragma unroll
        for (int t = 0; t < 2; ++t) {
            const int bn = colg * 32 + t * 16 + nl;
            v8bf lo = *(const v8bf*)&Bs[bn][8 * hh];
            v8bf hi = *(const v8bf*)&Bs[bn][16 + 8 * hh];
            v16bf bfrag;
            #pragma unroll
            for (int i = 0; i < 8; ++i) { bfrag[i] = lo[i]; bfrag[8 + i] = hi[i]; }
            if (t == 0) acc0 = __builtin_amdgcn_wmma_f32_16x16x32_bf16(false, a, false, bfrag, (short)0, acc0, false, false);
            else        acc1 = __builtin_amdgcn_wmma_f32_16x16x32_bf16(false, a, false, bfrag, (short)0, acc1, false, false);
        }
        __syncthreads();
    }
    // ---- epilogue: D[M=r+8h][N=nl] ----
    #pragma unroll
    for (int t = 0; t < 2; ++t) {
        v8f acc = t ? acc1 : acc0;
        const int col = n0 + colg * 32 + t * 16 + nl;
        if (col < N) {
            const float bv = bias ? bias[col] : 0.f;
            #pragma unroll
            for (int r = 0; r < 8; ++r) {
                const int row = m0 + rowg * 16 + r + 8 * hh;
                C[(size_t)row * N + col] = acc[r] + bv;
            }
        }
    }
}

// ---------------- flash attention: one wave = one 16-row Q tile of one (b,head) ----------------
// proj: [MTOT][288] = (k | v | q) per 96-ch group; out: [MTOT][96]
__global__ __launch_bounds__(128)
void attn_kernel(const float* __restrict__ proj, float* __restrict__ out)
{
    __shared__ __bf16 Ps[4][16][32];
    const int tid = threadIdx.x, lane = tid & 31, wv = tid >> 5;
    const int wid = blockIdx.x * 4 + wv;        // 0..4095
    const int qt = wid & 63;
    const int bh = wid >> 6;
    const int b = bh >> 2, hd = bh & 3;
    const int nl = lane & 15, hh = lane >> 4;
    const float scale = rsqrtf(24.0f);
    const size_t rowbase = (size_t)b * HW;
    const int coff = hd * 24;

    // Q fragment (rows qt*16+nl), dh padded 24->32; element i -> d=i+16h
    v16bf aq;
    {
        const float* qr = proj + (rowbase + qt * 16 + nl) * 288 + 192 + coff + 16 * hh;
        float qv[16];
        *(float4*)&qv[0] = *(const float4*)qr;
        *(float4*)&qv[4] = *(const float4*)(qr + 4);
        float4 z = {0.f, 0.f, 0.f, 0.f};
        *(float4*)&qv[8] = z; *(float4*)&qv[12] = z;
        if (hh == 0) {   // h=1 upper half is all pad (d=24..31)
            *(float4*)&qv[8]  = *(const float4*)(qr + 8);
            *(float4*)&qv[12] = *(const float4*)(qr + 12);
        }
        #pragma unroll
        for (int i = 0; i < 16; ++i) aq[i] = (__bf16)(qv[i] * scale);
    }

    v8f accA = {}, accB = {};
    float mrow[8], lrow[8];
    #pragma unroll
    for (int r = 0; r < 8; ++r) { mrow[r] = -1e30f; lrow[r] = 0.f; }

    for (int kc = 0; kc < HW; kc += 32) {
        // S = Q @ K^T for 32 keys (two 16-key N tiles); B-frag element i -> dim i+16h
        v8f S0 = {}, S1 = {};
        #pragma unroll
        for (int t = 0; t < 2; ++t) {
            const float* kr = proj + (rowbase + kc + t * 16 + nl) * 288 + coff + 16 * hh;
            float kv[16];
            *(float4*)&kv[0]  = *(const float4*)kr;        // stays in-row: coff+16h+15 <= 103
            *(float4*)&kv[4]  = *(const float4*)(kr + 4);
            *(float4*)&kv[8]  = *(const float4*)(kr + 8);
            *(float4*)&kv[12] = *(const float4*)(kr + 12);
            v16bf bk;
            #pragma unroll
            for (int i = 0; i < 16; ++i) {
                const int d = i + 16 * hh;
                bk[i] = (__bf16)((d < 24) ? kv[i] : 0.f);
            }
            if (t == 0) S0 = __builtin_amdgcn_wmma_f32_16x16x32_bf16(false, aq, false, bk, (short)0, S0, false, false);
            else        S1 = __builtin_amdgcn_wmma_f32_16x16x32_bf16(false, aq, false, bk, (short)0, S1, false, false);
        }
        // online softmax per row (rows r+8h; 16-lane halves share a row)
        float p0[8], p1[8];
        #pragma unroll
        for (int r = 0; r < 8; ++r) {
            float cm = red_max16(fmaxf(S0[r], S1[r]));
            const float mn = fmaxf(mrow[r], cm);
            const float corr = __expf(mrow[r] - mn);
            p0[r] = __expf(S0[r] - mn);
            p1[r] = __expf(S1[r] - mn);
            const float rs = red_sum16(p0[r] + p1[r]);
            lrow[r] = lrow[r] * corr + rs;
            mrow[r] = mn;
            accA[r] *= corr; accB[r] *= corr;
        }
        // C-frag -> A-frag relayout through per-wave LDS tile
        #pragma unroll
        for (int r = 0; r < 8; ++r) {
            Ps[wv][r + 8 * hh][nl]      = (__bf16)p0[r];
            Ps[wv][r + 8 * hh][16 + nl] = (__bf16)p1[r];
        }
        v16bf ap;
        {
            v8bf lo = *(const v8bf*)&Ps[wv][nl][8 * hh];
            v8bf hi = *(const v8bf*)&Ps[wv][nl][16 + 8 * hh];
            #pragma unroll
            for (int i = 0; i < 8; ++i) { ap[i] = lo[i]; ap[8 + i] = hi[i]; }
        }
        // P @ V (two dim tiles: 0..15 and 16..31(pad)); B-frag element i -> key (i<8?i:i+8)+8h
        #pragma unroll
        for (int t = 0; t < 2; ++t) {
            const int d = t * 16 + nl;
            v16bf bv;
            #pragma unroll
            for (int i = 0; i < 16; ++i) {
                const int key = kc + (i < 8 ? i : i + 8) + 8 * hh;
                bv[i] = (__bf16)((d < 24) ? proj[(rowbase + key) * 288 + 96 + coff + d] : 0.f);
            }
            if (t == 0) accA = __builtin_amdgcn_wmma_f32_16x16x32_bf16(false, ap, false, bv, (short)0, accA, false, false);
            else        accB = __builtin_amdgcn_wmma_f32_16x16x32_bf16(false, ap, false, bv, (short)0, accB, false, false);
        }
    }
    #pragma unroll
    for (int r = 0; r < 8; ++r) {
        const int row = qt * 16 + r + 8 * hh;
        const float inv = 1.0f / lrow[r];
        float* dst = out + (rowbase + row) * 96 + coff;
        dst[nl] = accA[r] * inv;
        if (nl + 16 < 24) dst[16 + nl] = accB[r] * inv;
    }
}

// ---------------- elementwise kernels ----------------
__global__ void transpose_in_kernel(const float* __restrict__ x, float* __restrict__ h0)
{   // x: NCHW (16,16,32,32); h0: [MTOT][8] from channels 8..15 (x_id)
    const int idx = blockIdx.x * 256 + threadIdx.x;
    if (idx >= MTOT * 8) return;
    const int m = idx >> 3, ci = idx & 7;
    const int b = m >> 10, sp = m & 1023;
    h0[idx] = x[((size_t)(b * 16 + 8 + ci) << 10) + sp];
}

__global__ void posenc_kernel(float* __restrict__ pe)
{
    const int idx = blockIdx.x * 256 + threadIdx.x;
    if (idx >= HW * 96) return;
    const int s = idx / 96, j = idx % 96;
    const int jj = (j < 48) ? j : j - 48;
    const float inv = __expf(-logf(10000.f) / 47.f * (float)jj);
    const float st = (float)s * inv;
    pe[idx] = (j < 48) ? __sinf(st) : __cosf(st);
}

__global__ void concat_elu_kernel(const float* __restrict__ in, float* __restrict__ out, int C)
{
    const int idx = blockIdx.x * 256 + threadIdx.x;
    if (idx >= MTOT * 2 * C) return;
    const int m = idx / (2 * C), c = idx % (2 * C);
    float v = in[(size_t)m * C + (c < C ? c : c - C)];
    if (c >= C) v = -v;
    out[idx] = (v > 0.f) ? v : expm1f(v);
}

__global__ void gate_res_kernel(const float* __restrict__ res, const float* __restrict__ g,
                                float* __restrict__ out, int C)
{   // out[m][c] = res[m][c] + g[m][c]*sigmoid(g[m][C+c]); g has stride 2C
    const int idx = blockIdx.x * 256 + threadIdx.x;
    if (idx >= MTOT * C) return;
    const int m = idx / C, c = idx % C;
    const float a  = g[(size_t)m * 2 * C + c];
    const float bb = g[(size_t)m * 2 * C + C + c];
    out[idx] = res[idx] + a * (1.f / (1.f + __expf(-bb)));
}

__global__ __launch_bounds__(128)
void layernorm_kernel(const float* __restrict__ in, const float* __restrict__ gamma,
                      const float* __restrict__ beta, float* __restrict__ out)
{
    __shared__ float red[128];
    const int m = blockIdx.x, c = threadIdx.x;
    const float v = (c < 96) ? in[(size_t)m * 96 + c] : 0.f;
    red[c] = v; __syncthreads();
    for (int s = 64; s > 0; s >>= 1) { if (c < s) red[c] += red[c + s]; __syncthreads(); }
    const float mu = red[0] / 96.f; __syncthreads();
    const float d = (c < 96) ? v - mu : 0.f;
    red[c] = d * d; __syncthreads();
    for (int s = 64; s > 0; s >>= 1) { if (c < s) red[c] += red[c + s]; __syncthreads(); }
    const float var = red[0] / 96.f;
    if (c < 96) out[(size_t)m * 96 + c] = d * rsqrtf(var + 1e-5f) * gamma[c] + beta[c];
}

__global__ void addpe_kernel(const float* __restrict__ in, const float* __restrict__ pe,
                             float* __restrict__ out)
{
    const int idx = blockIdx.x * 256 + threadIdx.x;
    if (idx >= MTOT * 96) return;
    const int m = idx / 96, c = idx % 96;
    out[idx] = in[idx] + pe[(m & 1023) * 96 + c];
}

// ---------------- final coupling: mixture CDF/PDF + ldj partials ----------------
__global__ __launch_bounds__(256)
void coupling_kernel(const float* __restrict__ x, const float* __restrict__ nn /*[MTOT][784]*/,
                     const float* __restrict__ rv, const float* __restrict__ rg,
                     float* __restrict__ d_out, float* __restrict__ partials)
{
    __shared__ float red[256];
    const int b   = blockIdx.x >> 5;    // 16 batches * 32 blocks
    const int blk = blockIdx.x & 31;
    const int e = blk * 256 + threadIdx.x;   // 0..8191 in batch
    const int ci = e >> 10, sp = e & 1023;
    const int m = (b << 10) + sp;
    const float* row = nn + (size_t)m * 784;

    const float s_ = row[ci];
    const float t_ = row[8 + ci];
    const float rwv = rg[ci] * rv[ci] * rsqrtf(rv[ci] * rv[ci]);
    const float a = rwv * tanhf(s_);
    const float xc = x[((size_t)(b * 16 + ci) << 10) + sp];

    float pi[32]; float pm = -1e30f;
    #pragma unroll
    for (int k = 0; k < 32; ++k) { pi[k] = row[(2 + k) * 8 + ci]; pm = fmaxf(pm, pi[k]); }
    float se = 0.f;
    #pragma unroll
    for (int k = 0; k < 32; ++k) se += __expf(pi[k] - pm);
    const float lse = pm + __logf(se);

    float lc[32], lp[32], maxc = -1e30f, maxp = -1e30f;
    #pragma unroll
    for (int k = 0; k < 32; ++k) {
        const float mu = row[(34 + k) * 8 + ci];
        const float ls = fmaxf(row[(66 + k) * 8 + ci], -7.f);
        const float z = (xc - mu) * __expf(-ls);
        const float lpi = pi[k] - lse;
        const float spz = fmaxf(z, 0.f) + log1pf(__expf(-fabsf(z)));  // softplus(z)
        lc[k] = lpi + (z - spz);                 // + log_sigmoid(z)
        lp[k] = lpi + z - ls - 2.f * spz;
        maxc = fmaxf(maxc, lc[k]); maxp = fmaxf(maxp, lp[k]);
    }
    float sc = 0.f, spd = 0.f;
    #pragma unroll
    for (int k = 0; k < 32; ++k) { sc += __expf(lc[k] - maxc); spd += __expf(lp[k] - maxp); }
    const float log_cdf = maxc + __logf(sc);
    const float log_pdf = maxp + __logf(spd);
    const float cdf = __expf(log_cdf);
    const float zt = -safe_logf(1.f / cdf - 1.f);
    const float sldj = -safe_logf(cdf) - safe_logf(1.f - cdf);
    const float contrib = log_pdf + sldj + a;

    const size_t oi = ((size_t)(b * 16 + ci) << 10) + sp;
    d_out[oi] = (zt + t_) * __expf(a);
    d_out[oi + 8192] = x[oi + 8192];   // x_id passthrough (channels 8..15)

    red[threadIdx.x] = contrib; __syncthreads();
    for (int s = 128; s > 0; s >>= 1) { if (threadIdx.x < s) red[threadIdx.x] += red[threadIdx.x + s]; __syncthreads(); }
    if (threadIdx.x == 0) partials[blockIdx.x] = red[0];
}

__global__ void ldj_final_kernel(const float* __restrict__ ldj_in, const float* __restrict__ partials,
                                 float* __restrict__ d_out)
{
    const int b = threadIdx.x;
    if (b < 16) {
        float s = ldj_in[b];
        for (int i = 0; i < 32; ++i) s += partials[b * 32 + i];
        d_out[262144 + b] = s;
    }
}

// ---------------- host orchestration ----------------
extern "C" void kernel_launch(void* const* d_in, const int* in_sizes, int n_in,
                              void* d_out, int out_size, void* d_ws, size_t ws_size,
                              hipStream_t stream)
{
    (void)in_sizes; (void)n_in; (void)out_size; (void)ws_size;
    auto F = [&](int i) { return (const float*)d_in[i]; };
    // input layout: 0:x 1:ldj 2:in_v 3:in_g 4:in_b 5:out_v 6:out_g 7:out_b 8:rescale_v 9:rescale_g
    // then 2 blocks x {conv_v,conv_g,conv_b,cgate_v,cgate_g,cgate_b, ain_v,ain_g, agate_v,agate_g,agate_b, ln1_g,ln1_b,ln2_g,ln2_b}
    const float* x_in = F(0);
    const float* ldj  = F(1);

    char* ws = (char*)d_ws; size_t off = 0;
    auto alloc = [&](size_t bytes) -> void* { void* p = ws + off; off = (off + bytes + 255) & ~(size_t)255; return p; };

    __bf16* Win  = (__bf16*)alloc((size_t)128 * 96 * 2);
    __bf16* Wout = (__bf16*)alloc((size_t)832 * 864 * 2);
    __bf16* Wc1[2], *Wg[2], *Wai[2], *Wag[2];
    for (int i = 0; i < 2; ++i) {
        Wc1[i] = (__bf16*)alloc((size_t)128 * 1728 * 2);
        Wg[i]  = (__bf16*)alloc((size_t)192 * 192 * 2);
        Wai[i] = (__bf16*)alloc((size_t)320 * 96 * 2);
        Wag[i] = (__bf16*)alloc((size_t)192 * 96 * 2);
    }
    float* h0     = (float*)alloc((size_t)MTOT * 8 * 4);
    float* pe     = (float*)alloc((size_t)HW * 96 * 4);
    float* xcur   = (float*)alloc((size_t)MTOT * 96 * 4);
    float* t1     = (float*)alloc((size_t)MTOT * 96 * 4);
    float* h96    = (float*)alloc((size_t)MTOT * 96 * 4);   // also reused as x+pos_enc
    float* attn_o = (float*)alloc((size_t)MTOT * 96 * 4);
    float* ce     = (float*)alloc((size_t)MTOT * 192 * 4);
    float* g192   = (float*)alloc((size_t)MTOT * 192 * 4);
    float* proj   = (float*)alloc((size_t)MTOT * 288 * 4);
    float* outbig = (float*)alloc((size_t)MTOT * 784 * 4);
    float* parts  = (float*)alloc(512 * 4);

    auto gemm = [&](const float* A, int Cin, int taps, const __bf16* Wt, const float* bias,
                    float* C, int N, int Kpad) {
        dim3 grid(MTOT / 64, (N + 63) / 64);
        gemm_conv_kernel<<<grid, 256, 0, stream>>>(A, Cin, taps, Wt, bias, C, N, Kpad);
    };

    // ---- weight-norm prep ----
    wnorm_kernel<<<128, 64, 0, stream>>>(F(2), F(3), Win, 96, 8, 9, 96);       // in conv
    wnorm_kernel<<<832, 64, 0, stream>>>(F(5), F(6), Wout, 784, 96, 9, 864);   // out conv
    for (int bi = 0; bi < 2; ++bi) {
        const int p = 10 + bi * 15;
        wnorm_kernel<<<128, 64, 0, stream>>>(F(p + 0), F(p + 1), Wc1[bi], 96, 192, 9, 1728);
        wnorm_kernel<<<192, 64, 0, stream>>>(F(p + 3), F(p + 4), Wg[bi], 192, 192, 1, 192);
        wnorm_kernel<<<320, 64, 0, stream>>>(F(p + 6), F(p + 7), Wai[bi], 288, 96, 1, 96);
        wnorm_kernel<<<192, 64, 0, stream>>>(F(p + 8), F(p + 9), Wag[bi], 192, 96, 1, 96);
    }

    // ---- forward ----
    transpose_in_kernel<<<512, 256, 0, stream>>>(x_in, h0);
    posenc_kernel<<<(HW * 96 + 255) / 256, 256, 0, stream>>>(pe);
    gemm(h0, 8, 9, Win, F(4), xcur, 96, 96);

    for (int bi = 0; bi < 2; ++bi) {
        const int p = 10 + bi * 15;
        // gated conv
        concat_elu_kernel<<<(MTOT * 192 + 255) / 256, 256, 0, stream>>>(xcur, ce, 96);
        gemm(ce, 192, 9, Wc1[bi], F(p + 2), h96, 96, 1728);
        concat_elu_kernel<<<(MTOT * 192 + 255) / 256, 256, 0, stream>>>(h96, ce, 96);
        gemm(ce, 192, 1, Wg[bi], F(p + 5), g192, 192, 192);
        gate_res_kernel<<<(MTOT * 96 + 255) / 256, 256, 0, stream>>>(xcur, g192, xcur, 96);
        // LN1
        layernorm_kernel<<<MTOT, 128, 0, stream>>>(xcur, F(p + 11), F(p + 12), t1);
        // gated attention
        addpe_kernel<<<(MTOT * 96 + 255) / 256, 256, 0, stream>>>(t1, pe, h96);
        gemm(h96, 96, 1, Wai[bi], nullptr, proj, 288, 96);
        attn_kernel<<<1024, 128, 0, stream>>>(proj, attn_o);
        gemm(attn_o, 96, 1, Wag[bi], F(p + 10), g192, 192, 96);
        gate_res_kernel<<<(MTOT * 96 + 255) / 256, 256, 0, stream>>>(t1, g192, xcur, 96);
        // LN2 (in-place safe)
        layernorm_kernel<<<MTOT, 128, 0, stream>>>(xcur, F(p + 13), F(p + 14), xcur);
    }

    gemm(xcur, 96, 9, Wout, F(7), outbig, 784, 864);

    coupling_kernel<<<512, 256, 0, stream>>>(x_in, outbig, F(8), F(9), (float*)d_out, parts);
    ldj_final_kernel<<<1, 32, 0, stream>>>(ldj, parts, (float*)d_out);
}
